// FLDAttention_45011257262250
// MI455X (gfx1250) — compile-verified, matrix operand order checked by reading, not run on hardware
//
#include <hip/hip_runtime.h>
#include <hip/hip_bf16.h>

// ---------------------------------------------------------------------------
// Problem constants (from the reference)
// ---------------------------------------------------------------------------
#define BB      16
#define PP      128
#define SS      4096
#define EE      512
#define HH      8
#define DD      64          // head dim
#define LATENT  128
#define SM_SCALE 0.125f     // 1/sqrt(64)
#define BIG_NEG  (-1.0e30f)

typedef __attribute__((ext_vector_type(16))) _Float16 v16h;
typedef __attribute__((ext_vector_type(8)))  _Float16 v8h;
typedef __attribute__((ext_vector_type(8)))  float    v8f;
typedef __attribute__((ext_vector_type(4)))  float    v4f;
typedef unsigned long long u64;

static __device__ __forceinline__ v8f wmma_f16(v16h a, v16h b, v8f c) {
  // D = A(16x32 f16) * B(32x16 f16) + C(16x16 f32)
  return __builtin_amdgcn_wmma_f32_16x16x32_f16(false, a, false, b,
                                                (short)0, c, false, false);
}
static __device__ __forceinline__ v16h cat8(v8h lo, v8h hi) {
  return __builtin_shufflevector(lo, hi, 0,1,2,3,4,5,6,7,8,9,10,11,12,13,14,15);
}
static __device__ __forceinline__ v16h ldfrag_h(const _Float16* p) {
  // halves [0..7] = p[0..7] (K 0..7 / 8..15), halves [8..15] = p[16..23]
  v8h lo = *(const v8h*)p;
  v8h hi = *(const v8h*)(p + 16);
  return cat8(lo, hi);
}
static __device__ __forceinline__ v16h ldfrag_f32(const float* p) {
  v4f f0 = *(const v4f*)(p);
  v4f f1 = *(const v4f*)(p + 4);
  v4f f2 = *(const v4f*)(p + 16);
  v4f f3 = *(const v4f*)(p + 20);
  v16h r;
#pragma unroll
  for (int i = 0; i < 4; ++i) {
    r[i]      = (_Float16)f0[i];
    r[4 + i]  = (_Float16)f1[i];
    r[8 + i]  = (_Float16)f2[i];
    r[12 + i] = (_Float16)f3[i];
  }
  return r;
}

// ---------------------------------------------------------------------------
// Kernel 1: transpose+convert fp32 weight [K][N] -> f16 WT[N][K]
// ---------------------------------------------------------------------------
__global__ void wt_kernel(const float* __restrict__ w, _Float16* __restrict__ wt,
                          int k, int n) {
  int idx = blockIdx.x * blockDim.x + threadIdx.x;
  if (idx >= k * n) return;
  int e = idx % k;           // contiguous in e -> coalesced writes
  int c = idx / k;
  wt[(size_t)c * k + e] = (_Float16)w[(size_t)e * n + c];
}

// ---------------------------------------------------------------------------
// Kernel 2: projection GEMM  Xp = X(rows x E) @ W(E x E) + bias
//   COLMAJOR==0 (Q,K): transposed scheme (M=n, N=row), store Xp[b,h,row,d]
//   COLMAJOR==1 (V)  : normal scheme   (M=row, N=n), store XpT[b,h,d,row]
// Template parameter => single straight-line v_wmma in the K-loop.
// Per wave: one 16x16 output tile, K-loop over E in steps of 32 (16 WMMAs).
// ---------------------------------------------------------------------------
template <int COLMAJOR>
__global__ __launch_bounds__(128) void proj_kernel(
    const float* __restrict__ X,
    const _Float16* __restrict__ WT,
    const float* __restrict__ bias,
    _Float16* __restrict__ Out,
    int Srows) {
  const int wave = threadIdx.x >> 5;
  const int lane = threadIdx.x & 31;
  const int lh   = lane & 15;
  const int eoff = (lane >= 16) ? 8 : 0;

  const int nbase   = (blockIdx.y * 4 + wave) * 16;
  const int rowbase = blockIdx.x * 16;

  const _Float16* pa0 = WT + (size_t)(nbase + lh) * EE + eoff;
  const float*    pb0 = X  + (size_t)(rowbase + lh) * EE + eoff;

  v8f acc = {};
#pragma unroll 4
  for (int e0 = 0; e0 < EE; e0 += 32) {
    v16h wfrag = ldfrag_h(pa0 + e0);
    v16h xfrag = ldfrag_f32(pb0 + e0);
    if (COLMAJOR)
      acc = wmma_f16(xfrag, wfrag, acc);   // M = row
    else
      acc = wmma_f16(wfrag, xfrag, acc);   // M = n (feature)
  }

  if (!COLMAJOR) {
    // C layout: lane -> fixed row g, regs r -> n = nbase+eoff+r (contiguous d)
    const int g  = rowbase + lh;
    const int b  = g / Srows, sr = g % Srows;
    const int n0 = nbase + eoff;
    const int h  = n0 / DD, d0 = n0 % DD;
    const float* bp = bias + n0;
    v8h o;
#pragma unroll
    for (int r = 0; r < 8; ++r) o[r] = (_Float16)(acc[r] + bp[r]);
    _Float16* dst = Out + (((size_t)(b * HH + h) * Srows + sr) * DD + d0);
    *(v8h*)dst = o;
  } else {
    // C layout: lane -> fixed n, regs r -> row = rowbase+eoff+r (contiguous s)
    const int n  = nbase + lh;
    const int h  = n / DD, d = n % DD;
    const float bv = bias[n];
    const int b   = rowbase / Srows;
    const int sr0 = (rowbase % Srows) + eoff;
    v8h o;
#pragma unroll
    for (int r = 0; r < 8; ++r) o[r] = (_Float16)(acc[r] + bv);
    _Float16* dst = Out + (((size_t)(b * HH + h) * DD + d) * Srows + sr0);
    *(v8h*)dst = o;
  }
}

// ---------------------------------------------------------------------------
// Kernel 3: flash attention per (b,h,p-tile) wave.
//  scoresT(16s x 16p) = Kp-tile(16s x 64d) x QpT(64d x 16p)  [2 WMMAs/tile]
//  online softmax: per-lane scalar stats (lane owns one p), 1 shfl_xor(16)
//  ctxT(64d x 16p) += VpT-tile(16d x 32s) x P^T(32s x 16p)    [4 WMMAs]
// Kp+VpT together are 128 MB -> resident in the 192 MB L2 across p-tiles.
// ---------------------------------------------------------------------------
__global__ __launch_bounds__(128) void attn_kernel(
    const _Float16* __restrict__ Qp,
    const _Float16* __restrict__ Kp,
    const _Float16* __restrict__ VpT,
    const unsigned char* __restrict__ mask,
    _Float16* __restrict__ Ctx) {
  const int wave = threadIdx.x >> 5;
  const int lane = threadIdx.x & 31;
  const int lh   = lane & 15;
  const int eoff = (lane >= 16) ? 8 : 0;

  const int bh = blockIdx.x;
  const int b  = bh / HH, h = bh % HH;
  const int pbase = (blockIdx.y * 4 + wave) * 16;

  const _Float16* qb = Qp + ((size_t)bh * PP + pbase) * DD;
  const _Float16* kb = Kp + (size_t)bh * SS * DD;
  const _Float16* vb = VpT + (size_t)bh * DD * SS;
  const unsigned char* mb = mask + (size_t)b * SS;

  // Q as B-fragments for both 32-wide d chunks (lane owns column p = pbase+lh)
  v16h Qf[2];
#pragma unroll
  for (int kk = 0; kk < 2; ++kk)
    Qf[kk] = ldfrag_h(qb + (size_t)lh * DD + kk * 32 + eoff);

  float m = BIG_NEG, l = 0.0f;
  v8f ctxT[4] = {};

  for (int s0 = 0; s0 < SS; s0 += 32) {
    // ---- scoresT for two 16-key tiles --------------------------------
    v8f sc[2];
#pragma unroll
    for (int st = 0; st < 2; ++st) {
      const _Float16* kp = kb + (size_t)(s0 + st * 16 + lh) * DD + eoff;
      v16h ka0 = ldfrag_h(kp);        // d 0..31 chunk
      v16h ka1 = ldfrag_h(kp + 32);   // d 32..63 chunk
      v8f t = {};
      t = wmma_f16(ka0, Qf[0], t);
      t = wmma_f16(ka1, Qf[1], t);
      sc[st] = t;
    }
    // ---- scale + key-padding mask (lane's 8 s values are contiguous) --
    u64 mk0 = *(const u64*)(mb + s0 + eoff);
    u64 mk1 = *(const u64*)(mb + s0 + 16 + eoff);
#pragma unroll
    for (int r = 0; r < 8; ++r) {
      float v0 = sc[0][r] * SM_SCALE;
      float v1 = sc[1][r] * SM_SCALE;
      if (((mk0 >> (8 * r)) & 0xffull) == 0) v0 = BIG_NEG;
      if (((mk1 >> (8 * r)) & 0xffull) == 0) v1 = BIG_NEG;
      sc[0][r] = v0; sc[1][r] = v1;
    }
    // ---- online softmax (row = p, per-lane scalar state) --------------
    float cmax = BIG_NEG;
#pragma unroll
    for (int r = 0; r < 8; ++r) cmax = fmaxf(cmax, fmaxf(sc[0][r], sc[1][r]));
    cmax = fmaxf(cmax, __shfl_xor(cmax, 16, 32));
    float mnew  = fmaxf(m, cmax);
    float alpha = __expf(m - mnew);
    float rsum  = 0.0f;
    v16h pf;                       // lands exactly in B-fragment layout
#pragma unroll
    for (int r = 0; r < 8; ++r) {
      float p0 = __expf(sc[0][r] - mnew);
      float p1 = __expf(sc[1][r] - mnew);
      rsum += p0 + p1;
      pf[r]     = (_Float16)p0;    // K = s 0..7  / 8..15
      pf[8 + r] = (_Float16)p1;    // K = s 16..23/ 24..31
    }
    rsum += __shfl_xor(rsum, 16, 32);
    l = l * alpha + rsum;
    m = mnew;
#pragma unroll
    for (int t = 0; t < 4; ++t)
#pragma unroll
      for (int r = 0; r < 8; ++r) ctxT[t][r] *= alpha;
    // ---- ctxT += VpT x P^T -------------------------------------------
#pragma unroll
    for (int dt = 0; dt < 4; ++dt) {
      const _Float16* vp = vb + (size_t)(dt * 16 + lh) * SS + s0 + eoff;
      v16h vf = ldfrag_h(vp);
      ctxT[dt] = wmma_f16(vf, pf, ctxT[dt]);
    }
    // ---- prefetch next K/V tiles into the cache hierarchy -------------
    if (s0 + 32 < SS) {
      __builtin_prefetch(kb + (size_t)(s0 + 32 + lh) * DD, 0, 1);
      __builtin_prefetch(vb + (size_t)lh * SS + s0 + 32, 0, 1);
    }
  }

  // ---- normalize and store ctx row-major (p, e) in f16 -----------------
  const float inv = 1.0f / l;
  _Float16* crow = Ctx + ((size_t)(b * PP + pbase + lh)) * EE + h * DD;
#pragma unroll
  for (int dt = 0; dt < 4; ++dt) {
    v8h o;
#pragma unroll
    for (int r = 0; r < 8; ++r) o[r] = (_Float16)(ctxT[dt][r] * inv);
    *(v8h*)(crow + dt * 16 + eoff) = o;
  }
}

// ---------------------------------------------------------------------------
// Kernel 4: output projection  Out = Ctx(B*P x E) @ wo(E x 128) + bo  (fp32)
// Transposed scheme: M = o, N = p  -> per-lane 8 contiguous fp32 stores.
// ---------------------------------------------------------------------------
__global__ __launch_bounds__(128) void outproj_kernel(
    const _Float16* __restrict__ Ctx,
    const _Float16* __restrict__ WOT,
    const float* __restrict__ bo,
    float* __restrict__ out) {
  const int wave = threadIdx.x >> 5;
  const int lane = threadIdx.x & 31;
  const int lh   = lane & 15;
  const int eoff = (lane >= 16) ? 8 : 0;

  const int obase   = (blockIdx.y * 4 + wave) * 16;
  const int rowbase = blockIdx.x * 16;

  const _Float16* pa0 = WOT + (size_t)(obase + lh) * EE + eoff;
  const _Float16* pb0 = Ctx + (size_t)(rowbase + lh) * EE + eoff;

  v8f acc = {};
#pragma unroll 4
  for (int e0 = 0; e0 < EE; e0 += 32) {
    v16h afrag = ldfrag_h(pa0 + e0);
    v16h bfrag = ldfrag_h(pb0 + e0);
    acc = wmma_f16(afrag, bfrag, acc);
  }
  const int g  = rowbase + lh;           // global p row
  const int o0 = obase + eoff;
  float* dst = out + (size_t)g * LATENT + o0;
  v4f lo4, hi4;
#pragma unroll
  for (int r = 0; r < 4; ++r) {
    lo4[r] = acc[r]     + bo[o0 + r];
    hi4[r] = acc[4 + r] + bo[o0 + 4 + r];
  }
  *(v4f*)dst       = lo4;
  *(v4f*)(dst + 4) = hi4;
}

// ---------------------------------------------------------------------------
// Host launcher. Workspace layout (all 256B aligned):
//   WTq/WTk/WTv f16[512][512], woT f16[128][512],
//   Qp f16[B][H][P][D], Kp f16[B][H][S][D], VpT f16[B][H][D][S],
//   Ctx f16[B][P][E]                                  (~134 MB total)
// ---------------------------------------------------------------------------
extern "C" void kernel_launch(void* const* d_in, const int* in_sizes, int n_in,
                              void* d_out, int out_size, void* d_ws, size_t ws_size,
                              hipStream_t stream) {
  const float* Q  = (const float*)d_in[0];
  const float* K  = (const float*)d_in[1];
  const float* V  = (const float*)d_in[2];
  const unsigned char* mask = (const unsigned char*)d_in[3];
  const float* wq = (const float*)d_in[4];
  const float* bq = (const float*)d_in[5];
  const float* wk = (const float*)d_in[6];
  const float* bk = (const float*)d_in[7];
  const float* wv = (const float*)d_in[8];
  const float* bv = (const float*)d_in[9];
  const float* wo = (const float*)d_in[10];
  const float* bo = (const float*)d_in[11];
  float* out = (float*)d_out;

  char* ws = (char*)d_ws;
  const size_t SZ_W  = (size_t)EE * EE * 2;        // 512 KB
  const size_t SZ_WO = (size_t)LATENT * EE * 2;    // 128 KB
  const size_t SZ_QP = (size_t)BB * PP * EE * 2;   // 2 MB
  const size_t SZ_KP = (size_t)BB * SS * EE * 2;   // 64 MB
  _Float16* WTQ = (_Float16*)(ws);
  _Float16* WTK = (_Float16*)(ws + SZ_W);
  _Float16* WTV = (_Float16*)(ws + 2 * SZ_W);
  _Float16* WOT = (_Float16*)(ws + 3 * SZ_W);
  _Float16* QP  = (_Float16*)(ws + 3 * SZ_W + SZ_WO);
  _Float16* KP  = (_Float16*)(ws + 3 * SZ_W + SZ_WO + SZ_QP);
  _Float16* VPT = (_Float16*)(ws + 3 * SZ_W + SZ_WO + SZ_QP + SZ_KP);
  _Float16* CTX = (_Float16*)(ws + 3 * SZ_W + SZ_WO + SZ_QP + 2 * SZ_KP);

  // 1) weight transpose+convert
  wt_kernel<<<(EE * EE + 255) / 256, 256, 0, stream>>>(wq, WTQ, EE, EE);
  wt_kernel<<<(EE * EE + 255) / 256, 256, 0, stream>>>(wk, WTK, EE, EE);
  wt_kernel<<<(EE * EE + 255) / 256, 256, 0, stream>>>(wv, WTV, EE, EE);
  wt_kernel<<<(EE * LATENT + 255) / 256, 256, 0, stream>>>(wo, WOT, EE, LATENT);

  // 2) projections (block = 4 waves, each wave one 16x16 tile; grid.y spans E/64)
  proj_kernel<0><<<dim3(BB * PP / 16, EE / 64), 128, 0, stream>>>(Q, WTQ, bq, QP, PP);
  proj_kernel<0><<<dim3(BB * SS / 16, EE / 64), 128, 0, stream>>>(K, WTK, bk, KP, SS);
  proj_kernel<1><<<dim3(BB * SS / 16, EE / 64), 128, 0, stream>>>(V, WTV, bv, VPT, SS);

  // 3) attention: one wave per (b,h,p-tile); 8 p-tiles -> grid.y=2 x 4 waves
  attn_kernel<<<dim3(BB * HH, PP / 64), 128, 0, stream>>>(QP, KP, VPT, mask, CTX);

  // 4) output projection + bias (fp32 out)
  outproj_kernel<<<dim3(BB * PP / 16, LATENT / 64), 128, 0, stream>>>(CTX, WOT, bo, out);
}
